// TransformerDecoderLayer_56375740727936
// MI455X (gfx1250) — matrix operand order repaired and checked by reading, not compiled, not run on hardware
//
#include <hip/hip_runtime.h>
#include <hip/hip_bf16.h>
#include <stdint.h>

// ---------------- problem constants ----------------
#define BB   2
#define TT   1024
#define SS   1024
#define NN   512
#define DD   1024
#define HH   16
#define HSZ  64
#define FF_  4096
#define RELSZ 33   // 2*16+1

typedef float  v8f   __attribute__((ext_vector_type(8)));
typedef __bf16 v16bf __attribute__((ext_vector_type(16)));
typedef unsigned int v4u __attribute__((ext_vector_type(4)));
typedef int v4i __attribute__((ext_vector_type(4)));
typedef int v8i __attribute__((ext_vector_type(8)));

#if __has_builtin(__builtin_amdgcn_tensor_load_to_lds) && __has_builtin(__builtin_amdgcn_s_wait_tensorcnt)
#define HAVE_TDM 1
#else
#define HAVE_TDM 0
#endif

struct BfPack { uint4 x; uint4 y; };

// A-fragment (16x32 bf16, MxK): lanes<16 rows 0..15 with K = kb..kb+7 & kb+16..kb+23 (kb=0);
// lanes>=16 same rows, kb=8.  Pass kb = sliceBase + (half?8:0).
static __device__ __forceinline__ v16bf ldA(const __bf16* row, int kb) {
    BfPack u;
    u.x = *(const uint4*)(row + kb);
    u.y = *(const uint4*)(row + kb + 16);
    return __builtin_bit_cast(v16bf, u);
}
// B-fragment (32x16 bf16, KxN) from [n][k] storage: lanes<16 n=lane K = kb..kb+7 & kb+16..kb+23 (kb=0);
// lanes>=16 kb=8.  Pass kb = sliceBase + (half?8:0).
static __device__ __forceinline__ v16bf ldB(const __bf16* row, int kb) {
    BfPack u;
    u.x = *(const uint4*)(row + kb);
    u.y = *(const uint4*)(row + kb + 16);
    return __builtin_bit_cast(v16bf, u);
}
static __device__ __forceinline__ v8f wmma_bf16(v16bf a, v16bf b, v8f c) {
    return __builtin_amdgcn_wmma_f32_16x16x32_bf16(false, a, false, b, (short)0, c, false, false);
}
#define V8ZERO (v8f{0.f,0.f,0.f,0.f,0.f,0.f,0.f,0.f})

#if HAVE_TDM
// Tensor Data Mover: 2-D bf16 tile (tile1 rows of tile0 elements, row stride stride0 elems)
// global -> LDS rows packed consecutively.  D# per CDNA5 ISA 8.3/8.4.
static __device__ __forceinline__ void tdm_load_2d(unsigned lds_addr, unsigned long long gaddr,
                                                   unsigned td0, unsigned td1,
                                                   unsigned tile0, unsigned tile1,
                                                   unsigned long long stride0) {
    v4u g0;
    g0.x = 1u;                                         // count=1 (valid user descriptor)
    g0.y = lds_addr;                                   // lds_addr [63:32]
    g0.z = (unsigned)(gaddr & 0xffffffffu);            // global_addr[31:0]
    g0.w = (unsigned)((gaddr >> 32) & 0x1ffffffu) | (2u << 30);  // ga[56:32] | type=2
    v8i g1;
    g1[0] = (int)(1u << 16);                           // workgroup_mask=0, data_size=1 (2B)
    g1[1] = (int)((td0 & 0xffffu) << 16);              // tensor_dim0[15:0]
    g1[2] = (int)(((td0 >> 16) & 0xffffu) | ((td1 & 0xffffu) << 16));
    g1[3] = (int)(((td1 >> 16) & 0xffffu) | ((tile0 & 0xffffu) << 16));
    g1[4] = (int)(tile1 & 0xffffu);                    // tile_dim1 | tile_dim2=0
    g1[5] = (int)(unsigned)(stride0 & 0xffffffffu);    // tensor_dim0_stride[31:0]
    g1[6] = (int)(unsigned)((stride0 >> 32) & 0xffffu);
    g1[7] = 0;
    v4i z4 = {0, 0, 0, 0};
#if defined(__clang_major__) && (__clang_major__ >= 23)
    v8i z8 = {0, 0, 0, 0, 0, 0, 0, 0};
    __builtin_amdgcn_tensor_load_to_lds(g0, g1, z4, z4, z8, 0);
#else
    __builtin_amdgcn_tensor_load_to_lds(g0, g1, z4, z4, 0);
#endif
}
#endif

// ---------------- LayerNorm (row = D = 1024) ----------------
__global__ __launch_bounds__(256) void ln_kernel(const float* __restrict__ x,
                                                 const float* __restrict__ g,
                                                 const float* __restrict__ b,
                                                 float* __restrict__ y) {
    const int row = blockIdx.x, tid = threadIdx.x;
    __shared__ float red[256];
    const float* xr = x + (size_t)row * DD;
    float s = 0.f, ss = 0.f;
    #pragma unroll
    for (int it = 0; it < 4; ++it) { float v = xr[tid + it * 256]; s += v; ss += v * v; }
    red[tid] = s; __syncthreads();
    for (int o = 128; o > 0; o >>= 1) { if (tid < o) red[tid] += red[tid + o]; __syncthreads(); }
    const float mean = red[0] * (1.f / DD);
    __syncthreads();
    red[tid] = ss; __syncthreads();
    for (int o = 128; o > 0; o >>= 1) { if (tid < o) red[tid] += red[tid + o]; __syncthreads(); }
    const float var = red[0] * (1.f / DD) - mean * mean;
    const float inv = rsqrtf(var + 1e-5f);
    #pragma unroll
    for (int it = 0; it < 4; ++it) {
        int i = tid + it * 256;
        y[(size_t)row * DD + i] = (xr[i] - mean) * inv * g[i] + b[i];
    }
}

// ---------------- relative-K pre-reduction: R[bh,t,d] = q(scaled,bf16) . rel_k[d] ----------------
__global__ void relscore_kernel(const __bf16* __restrict__ q,
                                const float* __restrict__ rel_k,
                                float* __restrict__ R, int total) {
    int idx = blockIdx.x * blockDim.x + threadIdx.x;
    if (idx >= total) return;
    int d  = idx % RELSZ;
    int bt = idx / RELSZ;
    const __bf16* qp = q + (size_t)bt * HSZ;
    const float*  rp = rel_k + (size_t)d * HSZ;
    float s = 0.f;
    #pragma unroll 8
    for (int j = 0; j < HSZ; ++j) s += (float)qp[j] * rp[j];
    R[idx] = s;
}

// ---------------- WMMA GEMM, 128x128 tile, 256 threads (8 waves, 32x64 each) ----------------
// LDS double-buffered: one barrier per k-step; global loads overlap WMMA.
// A [M,K] f32, W [K,N] f32.  outmode 0: f32 Y=alpha*(AW+b)+resid, relu opt.
// outmode 1: bf16 head-major  Y[((bb*H+h)*L + t)*64 + hs]
// outmode 2: bf16 head-transposed (for V)  Y[((bb*H+h)*64 + hs)*L + t]
__global__ __launch_bounds__(256) void gemm_kernel(const float* __restrict__ A,
                                                   const float* __restrict__ W,
                                                   const float* __restrict__ bias,
                                                   void* __restrict__ Yv,
                                                   const float* __restrict__ resid,
                                                   int M, int N, int K, int Lrows,
                                                   float alpha, int relu, int outmode) {
    const int tid = threadIdx.x, w = tid >> 5, lane = tid & 31;
    const int lrow = lane & 15, half = lane >> 4;
    const int wm = w >> 1, wn = w & 1;                 // 4x2 wave grid
    const int mBase = blockIdx.y * 128, nBase = blockIdx.x * 128;

    __shared__ __attribute__((aligned(16))) __bf16 As[2][128][32];
    __shared__ __attribute__((aligned(16))) __bf16 Bs[2][128][32];   // [n][k]

    v8f acc[2][4];
    #pragma unroll
    for (int mt = 0; mt < 2; ++mt)
        #pragma unroll
        for (int nt = 0; nt < 4; ++nt) acc[mt][nt] = V8ZERO;

    const int kbA = half ? 8 : 0;
    const int kbB = half ? 8 : 0;

    // per-thread staging indices
    const int ar = tid >> 1, ac = (tid & 1) << 4;       // A: 2 float4 per row-half? (see loads)
    float4 ra[4];
    float  rb[4][4];
    auto loadAB = [&](int k0) {
        #pragma unroll
        for (int it = 0; it < 4; ++it) {
            int j = tid + it * 256;
            int r = j >> 3, c = (j & 7) << 2;           // A: row r, k-cols c..c+3
            ra[it] = *(const float4*)(A + (size_t)(mBase + r) * K + k0 + c);
            int n = j & 127, kg = (j >> 7) << 2;        // B: col n, k-rows kg..kg+3
            const float* wp = W + (size_t)(k0 + kg) * N + nBase + n;
            rb[it][0] = wp[0];
            rb[it][1] = wp[N];
            rb[it][2] = wp[2 * (size_t)N];
            rb[it][3] = wp[3 * (size_t)N];
        }
    };
    auto stageAB = [&](int buf) {
        #pragma unroll
        for (int it = 0; it < 4; ++it) {
            int j = tid + it * 256;
            int r = j >> 3, c = (j & 7) << 2;
            As[buf][r][c]     = (__bf16)ra[it].x;
            As[buf][r][c + 1] = (__bf16)ra[it].y;
            As[buf][r][c + 2] = (__bf16)ra[it].z;
            As[buf][r][c + 3] = (__bf16)ra[it].w;
            int n = j & 127, kg = (j >> 7) << 2;
            __bf16* bp = &Bs[buf][n][kg];               // 4 k-consecutive -> one b64 store
            bp[0] = (__bf16)rb[it][0];
            bp[1] = (__bf16)rb[it][1];
            bp[2] = (__bf16)rb[it][2];
            bp[3] = (__bf16)rb[it][3];
        }
    };

    const int nsteps = K >> 5;
    loadAB(0);
    stageAB(0);
    __syncthreads();
    for (int s = 0; s < nsteps; ++s) {
        const int cur = s & 1;
        const bool more = (s + 1 < nsteps);
        if (more) loadAB((s + 1) << 5);                 // global loads overlap WMMA below

        v16bf a0 = ldA(&As[cur][wm * 32 + lrow][0], kbA);
        v16bf a1 = ldA(&As[cur][wm * 32 + 16 + lrow][0], kbA);
        v16bf b0 = ldB(&Bs[cur][wn * 64 + lrow][0], kbB);
        v16bf b1 = ldB(&Bs[cur][wn * 64 + 16 + lrow][0], kbB);
        v16bf b2 = ldB(&Bs[cur][wn * 64 + 32 + lrow][0], kbB);
        v16bf b3 = ldB(&Bs[cur][wn * 64 + 48 + lrow][0], kbB);
        acc[0][0] = wmma_bf16(a0, b0, acc[0][0]);
        acc[0][1] = wmma_bf16(a0, b1, acc[0][1]);
        acc[0][2] = wmma_bf16(a0, b2, acc[0][2]);
        acc[0][3] = wmma_bf16(a0, b3, acc[0][3]);
        acc[1][0] = wmma_bf16(a1, b0, acc[1][0]);
        acc[1][1] = wmma_bf16(a1, b1, acc[1][1]);
        acc[1][2] = wmma_bf16(a1, b2, acc[1][2]);
        acc[1][3] = wmma_bf16(a1, b3, acc[1][3]);

        if (more) stageAB(cur ^ 1);                     // fill idle buffer
        __syncthreads();                                // single barrier per k-step
    }

    float*  Yf = (float*)Yv;
    __bf16* Yb = (__bf16*)Yv;
    #pragma unroll
    for (int mt = 0; mt < 2; ++mt)
    #pragma unroll
    for (int nt = 0; nt < 4; ++nt)
    #pragma unroll
    for (int i = 0; i < 8; ++i) {
        int row = mBase + wm * 32 + mt * 16 + i + half * 8;
        int col = nBase + wn * 64 + nt * 16 + lrow;
        float v = (acc[mt][nt][i] + bias[col]) * alpha;
        if (outmode == 0) {
            if (resid) v += resid[(size_t)row * N + col];
            if (relu)  v = fmaxf(v, 0.f);
            Yf[(size_t)row * N + col] = v;
        } else {
            int bb = row / Lrows, t = row % Lrows;
            int h = col >> 6, hs = col & 63;
            if (outmode == 1)
                Yb[(((size_t)(bb * HH + h)) * Lrows + t) * HSZ + hs] = (__bf16)v;
            else
                Yb[(((size_t)(bb * HH + h)) * HSZ + hs) * Lrows + t] = (__bf16)v;
        }
    }
}

// ---------------- flash attention ----------------
// Q,K bf16 [bh][L][64]; V bf16 head-transposed [bh][64][L].  64 q rows / WG, 16 / wave.
// K/V tiles staged to LDS with the Tensor Data Mover (double buffered), wave0 issues.
template <bool REL>
__global__ __launch_bounds__(128) void flash_kernel(const __bf16* __restrict__ Qp,
                                                    const __bf16* __restrict__ Kp,
                                                    const __bf16* __restrict__ Vp,
                                                    const unsigned char* __restrict__ mask,
                                                    int maskRowwise,
                                                    const float* __restrict__ Rrel,
                                                    const float* __restrict__ relv,
                                                    float* __restrict__ ctx,
                                                    float* __restrict__ mOut,
                                                    float* __restrict__ lOut,
                                                    int Lq, int Lk) {
    const int tid = threadIdx.x, w = tid >> 5, lane = tid & 31;
    const int lrow = lane & 15, half = lane >> 4;
    const int bh = blockIdx.y, b = bh >> 4, h = bh & 15;
    const int q0 = blockIdx.x * 64 + w * 16;

    __shared__ __attribute__((aligned(16))) __bf16 Ks[2][32][64];   // [key][hs]
    __shared__ __attribute__((aligned(16))) __bf16 Vs[2][64][32];   // [hs][key]
    __shared__ __attribute__((aligned(16))) __bf16 Ps[4][16][32];
    constexpr int RELN = REL ? (2 * 4 * 16 * RELSZ + RELSZ * HSZ) : 1;
    __shared__ float relbuf[RELN];
    float* Rs  = relbuf;
    float* Wh  = relbuf + 4 * 16 * RELSZ;
    float* RVs = relbuf + 2 * 4 * 16 * RELSZ;

    // Q fragments straight from global (bf16, b128 loads)
    const __bf16* qrow = Qp + ((size_t)bh * Lq + q0 + lrow) * HSZ;
    const int kbA = half ? 8 : 0;
    const int kbB = half ? 8 : 0;
    const v16bf qa0 = ldA(qrow, 0 + kbA);
    const v16bf qa1 = ldA(qrow, 32 + kbA);

    if constexpr (REL) {
        for (int i = lane; i < 16 * RELSZ; i += 32) {
            int r = i / RELSZ, d = i % RELSZ;
            Rs[(w * 16 + r) * RELSZ + d] = Rrel[((size_t)bh * Lq + q0 + r) * RELSZ + d];
            Wh[(w * 16 + r) * RELSZ + d] = 0.f;
        }
        #pragma unroll
        for (int it = 0; it < 17; ++it) {
            int i = tid + it * 128;
            if (i < RELSZ * HSZ) RVs[i] = relv[i];
        }
    }

    const size_t kvbase = (size_t)bh * Lk * HSZ;      // elements
#if HAVE_TDM
    const unsigned long long kAddr = (unsigned long long)(uintptr_t)Kp + kvbase * 2ull;
    const unsigned long long vAddr = (unsigned long long)(uintptr_t)Vp + (unsigned long long)bh * HSZ * (unsigned long long)Lk * 2ull;
#endif
    auto stage_tiles = [&](int kt2, int buf) {
#if HAVE_TDM
        if (w == 0) {
            tdm_load_2d((unsigned)(uintptr_t)&Ks[buf][0][0],
                        kAddr + (unsigned long long)kt2 * HSZ * 2ull,
                        /*td0*/HSZ, /*td1*/(unsigned)Lk, /*tile0*/HSZ, /*tile1*/32, /*stride0*/HSZ);
            tdm_load_2d((unsigned)(uintptr_t)&Vs[buf][0][0],
                        vAddr + (unsigned long long)kt2 * 2ull,
                        /*td0*/(unsigned)Lk, /*td1*/HSZ, /*tile0*/32, /*tile1*/HSZ,
                        /*stride0*/(unsigned long long)Lk);
        }
#else
        #pragma unroll
        for (int it = 0; it < 2; ++it) {
            int j = tid + it * 128;
            int kr = j >> 3, c = (j & 7) << 3;
            *(uint4*)&Ks[buf][kr][c] = *(const uint4*)(Kp + kvbase + (size_t)(kt2 + kr) * HSZ + c);
            int r = j >> 2, cv = (j & 3) << 3;
            *(uint4*)&Vs[buf][r][cv] = *(const uint4*)(Vp + (size_t)bh * HSZ * Lk + (size_t)r * Lk + kt2 + cv);
        }
#endif
    };

    v8f O0 = V8ZERO, O1 = V8ZERO, O2 = V8ZERO, O3 = V8ZERO;
    float mrow[8], lsum[8];
    #pragma unroll
    for (int i = 0; i < 8; ++i) { mrow[i] = -__builtin_inff(); lsum[i] = 0.f; }

    stage_tiles(0, 0);
    for (int kt = 0; kt < Lk; kt += 32) {
        const int cur = (kt >> 5) & 1;
        if (kt + 32 < Lk) {
            stage_tiles(kt + 32, cur ^ 1);
#if HAVE_TDM
            if (w == 0) __builtin_amdgcn_s_wait_tensorcnt(2);
#endif
        } else {
#if HAVE_TDM
            if (w == 0) __builtin_amdgcn_s_wait_tensorcnt(0);
#endif
        }
        __syncthreads();

        v8f s0 = V8ZERO, s1 = V8ZERO;
        #pragma unroll
        for (int s = 0; s < 2; ++s) {
            v16bf qa = s ? qa1 : qa0;
            v16bf bk0 = ldB(&Ks[cur][0 + lrow][0],  s * 32 + kbB);
            v16bf bk1 = ldB(&Ks[cur][16 + lrow][0], s * 32 + kbB);
            s0 = wmma_bf16(qa, bk0, s0);
            s1 = wmma_bf16(qa, bk1, s1);
        }

        #pragma unroll
        for (int i = 0; i < 8; ++i) {
            const int rl   = i + half * 8;
            const int rowg = q0 + rl;
            const int key0 = kt + lrow;
            const int key1 = kt + 16 + lrow;
            float e0 = s0[i], e1 = s1[i];
            int d0 = 0, d1 = 0;
            if constexpr (REL) {
                d0 = min(max(key0 - rowg, -16), 16) + 16;
                d1 = min(max(key1 - rowg, -16), 16) + 16;
                e0 += Rs[(w * 16 + rl) * RELSZ + d0];
                e1 += Rs[(w * 16 + rl) * RELSZ + d1];
            }
            if (mask) {
                size_t mi0 = maskRowwise ? ((size_t)(b * Lq + rowg)) * Lk + key0
                                         : (size_t)b * Lk + key0;
                size_t mi1 = maskRowwise ? ((size_t)(b * Lq + rowg)) * Lk + key1
                                         : (size_t)b * Lk + key1;
                e0 = mask[mi0] ? e0 : -__builtin_inff();
                e1 = mask[mi1] ? e1 : -__builtin_inff();
            }
            float mx = fmaxf(e0, e1);
            mx = fmaxf(mx, __shfl_xor(mx, 1));
            mx = fmaxf(mx, __shfl_xor(mx, 2));
            mx = fmaxf(mx, __shfl_xor(mx, 4));
            mx = fmaxf(mx, __shfl_xor(mx, 8));
            const float nm = fmaxf(mrow[i], mx);
            const float sc = (mrow[i] == -__builtin_inff()) ? 0.f : __expf(mrow[i] - nm);
            const float p0 = (nm == -__builtin_inff()) ? 0.f : __expf(e0 - nm);
            const float p1 = (nm == -__builtin_inff()) ? 0.f : __expf(e1 - nm);
            float rs = p0 + p1;
            rs += __shfl_xor(rs, 1);
            rs += __shfl_xor(rs, 2);
            rs += __shfl_xor(rs, 4);
            rs += __shfl_xor(rs, 8);
            lsum[i] = lsum[i] * sc + rs;
            mrow[i] = nm;
            O0[i] *= sc; O1[i] *= sc; O2[i] *= sc; O3[i] *= sc;
            Ps[w][rl][lrow]      = (__bf16)p0;
            Ps[w][rl][16 + lrow] = (__bf16)p1;
            if constexpr (REL) {
                atomicAdd(&Wh[(w * 16 + rl) * RELSZ + d0], p0);
                atomicAdd(&Wh[(w * 16 + rl) * RELSZ + d1], p1);
            }
        }

        // P (16x32) x V (32x64)
        v16bf pa = ldA(&Ps[w][lrow][0], kbA);
        v16bf v0 = ldB(&Vs[cur][0 + lrow][0],  kbB);
        v16bf v1 = ldB(&Vs[cur][16 + lrow][0], kbB);
        v16bf v2 = ldB(&Vs[cur][32 + lrow][0], kbB);
        v16bf v3 = ldB(&Vs[cur][48 + lrow][0], kbB);
        O0 = wmma_bf16(pa, v0, O0);
        O1 = wmma_bf16(pa, v1, O1);
        O2 = wmma_bf16(pa, v2, O2);
        O3 = wmma_bf16(pa, v3, O3);
        __syncthreads();
    }

    #pragma unroll
    for (int i = 0; i < 8; ++i) {
        const int rl   = i + half * 8;
        const int rowg = q0 + rl;
        float o0 = O0[i], o1 = O1[i], o2 = O2[i], o3 = O3[i];
        if constexpr (REL) {
            for (int d = 0; d < RELSZ; ++d) {
                const float wv = Wh[(w * 16 + rl) * RELSZ + d];
                o0 += wv * RVs[d * HSZ + 0 * 16 + lrow];
                o1 += wv * RVs[d * HSZ + 1 * 16 + lrow];
                o2 += wv * RVs[d * HSZ + 2 * 16 + lrow];
                o3 += wv * RVs[d * HSZ + 3 * 16 + lrow];
            }
        }
        const float inv = 1.f / lsum[i];
        const size_t cb = ((size_t)b * Lq + rowg) * DD + (size_t)h * HSZ;
        ctx[cb + 0 * 16 + lrow] = o0 * inv;
        ctx[cb + 1 * 16 + lrow] = o1 * inv;
        ctx[cb + 2 * 16 + lrow] = o2 * inv;
        ctx[cb + 3 * 16 + lrow] = o3 * inv;
        if (mOut && lrow == 0) {
            mOut[(size_t)bh * Lq + rowg] = mrow[i];
            lOut[(size_t)bh * Lq + rowg] = lsum[i];
        }
    }
}

// ---------------- CA attention-weight head-mean ----------------
__global__ __launch_bounds__(128) void ca_probs_kernel(const __bf16* __restrict__ Qp,
                                                       const __bf16* __restrict__ Kp,
                                                       const unsigned char* __restrict__ mask,
                                                       const float* __restrict__ mIn,
                                                       const float* __restrict__ lIn,
                                                       float* __restrict__ caw,
                                                       int Lq, int Lk) {
    const int tid = threadIdx.x, w = tid >> 5, lane = tid & 31;
    const int lrow = lane & 15, half = lane >> 4;
    const int b = blockIdx.z;
    const int q0 = blockIdx.y * 16;
    const int k0 = blockIdx.x * 128;

    __shared__ __attribute__((aligned(16))) __bf16 Ks[128][64];

    v8f acc0 = V8ZERO, acc1 = V8ZERO;
    const int kbA = half ? 8 : 0;
    const int kbB = half ? 8 : 0;

    for (int h = 0; h < HH; ++h) {
        __syncthreads();
        const size_t kb = ((size_t)(b * HH + h) * Lk + k0) * HSZ;
#if HAVE_TDM
        if (w == 0) {
            tdm_load_2d((unsigned)(uintptr_t)&Ks[0][0],
                        (unsigned long long)(uintptr_t)Kp + (unsigned long long)kb * 2ull,
                        /*td0*/HSZ, /*td1*/(unsigned)Lk, /*tile0*/HSZ, /*tile1*/128, /*stride0*/HSZ);
            __builtin_amdgcn_s_wait_tensorcnt(0);
        }
#else
        #pragma unroll
        for (int it = 0; it < 8; ++it) {
            int j = tid + it * 128;
            int r = j >> 3, c = (j & 7) << 3;
            *(uint4*)&Ks[r][c] = *(const uint4*)(Kp + kb + (size_t)r * HSZ + c);
        }
#endif
        __syncthreads();

        const __bf16* qrow = Qp + ((size_t)(b * HH + h) * Lq + q0 + lrow) * HSZ;
        v8f s0 = V8ZERO, s1 = V8ZERO;
        #pragma unroll
        for (int s = 0; s < 2; ++s) {
            v16bf qa = ldA(qrow, s * 32 + kbA);
            v16bf b0 = ldB(&Ks[w * 32 + lrow][0],      s * 32 + kbB);
            v16bf b1 = ldB(&Ks[w * 32 + 16 + lrow][0], s * 32 + kbB);
            s0 = wmma_bf16(qa, b0, s0);
            s1 = wmma_bf16(qa, b1, s1);
        }
        #pragma unroll
        for (int i = 0; i < 8; ++i) {
            const int rl = i + half * 8, rowg = q0 + rl;
            const float mv = mIn[(size_t)(b * HH + h) * Lq + rowg];
            const float invl = 1.f / lIn[(size_t)(b * HH + h) * Lq + rowg];
            const int key0 = k0 + w * 32 + lrow;
            const int key1 = key0 + 16;
            acc0[i] += mask[(size_t)b * Lk + key0] ? __expf(s0[i] - mv) * invl : 0.f;
            acc1[i] += mask[(size_t)b * Lk + key1] ? __expf(s1[i] - mv) * invl : 0.f;
        }
    }
    #pragma unroll
    for (int i = 0; i < 8; ++i) {
        const int rl = i + half * 8, rowg = q0 + rl;
        const int key0 = k0 + w * 32 + lrow;
        caw[((size_t)b * Lq + rowg) * Lk + key0]      = acc0[i] * (1.f / HH);
        caw[((size_t)b * Lq + rowg) * Lk + key0 + 16] = acc1[i] * (1.f / HH);
    }
}

// ---------------- host orchestration ----------------
extern "C" void kernel_launch(void* const* d_in, const int* in_sizes, int n_in,
                              void* d_out, int out_size, void* d_ws, size_t ws_size,
                              hipStream_t stream) {
    const float* trg  = (const float*)d_in[0];
    const float* code = (const float*)d_in[1];
    const float* ast  = (const float*)d_in[2];
    const unsigned char* src_mask  = (const unsigned char*)d_in[3];
    const unsigned char* node_mask = (const unsigned char*)d_in[4];
    const unsigned char* trg_mask  = (const unsigned char*)d_in[5];
    const float *sa_qW = (const float*)d_in[6],  *sa_qb = (const float*)d_in[7];
    const float *sa_kW = (const float*)d_in[8],  *sa_kb = (const float*)d_in[9];
    const float *sa_vW = (const float*)d_in[10], *sa_vb = (const float*)d_in[11];
    const float *sa_oW = (const float*)d_in[12], *sa_ob = (const float*)d_in[13];
    const float *ga_qW = (const float*)d_in[14], *ga_qb = (const float*)d_in[15];
    const float *ga_kW = (const float*)d_in[16], *ga_kb = (const float*)d_in[17];
    const float *ga_vW = (const float*)d_in[18], *ga_vb = (const float*)d_in[19];
    const float *ga_oW = (const float*)d_in[20], *ga_ob = (const float*)d_in[21];
    const float *ca_qW = (const float*)d_in[22], *ca_qb = (const float*)d_in[23];
    const float *ca_kW = (const float*)d_in[24], *ca_kb = (const float*)d_in[25];
    const float *ca_vW = (const float*)d_in[26], *ca_vb = (const float*)d_in[27];
    const float *ca_oW = (const float*)d_in[28], *ca_ob = (const float*)d_in[29];
    const float *rel_k = (const float*)d_in[30], *rel_v = (const float*)d_in[31];
    const float *ffW1 = (const float*)d_in[32], *ffb1 = (const float*)d_in[33];
    const float *ffW2 = (const float*)d_in[34], *ffb2 = (const float*)d_in[35];
    const float *ln1g = (const float*)d_in[36], *ln1b = (const float*)d_in[37];
    const float *ln2g = (const float*)d_in[38], *ln2b = (const float*)d_in[39];
    const float *ln3g = (const float*)d_in[40], *ln3b = (const float*)d_in[41];
    const float *ffg  = (const float*)d_in[42], *ffb  = (const float*)d_in[43];

    const size_t MB = 1024 * 1024;
    char* wsb = (char*)d_ws;
    float*  x   = (float*)(wsb + 0 * MB);     // [B*T, D] running residual   8 MB
    float*  xn  = (float*)(wsb + 8 * MB);     // LN output                   8 MB
    float*  ctx = (float*)(wsb + 16 * MB);    // [B*T, D]                    8 MB
    float*  ffh = (float*)(wsb + 24 * MB);    // [B*T, FF]                  32 MB
    __bf16* qb  = (__bf16*)(wsb + 56 * MB);   // [B,H,L,64] bf16             4 MB
    __bf16* kb  = (__bf16*)(wsb + 60 * MB);   //                             4 MB
    __bf16* vb  = (__bf16*)(wsb + 64 * MB);   // head-transposed [B,H,64,L]  4 MB
    float*  Rr  = (float*)(wsb + 68 * MB);    // [B,H,T,33]                ~4.2 MB
    float*  mca = (float*)(wsb + 73 * MB);    // [B,H,T]
    float*  lca = mca + (size_t)BB * HH * TT;

    float* out = (float*)d_out;
    float* caw = out + (size_t)BB * TT * DD;

    const dim3 blkG(256), blkF(128);
    const dim3 gD(DD / 128, (BB * TT) / 128);
    const float ISQ = 0.125f;

    // ===== self attention =====
    ln_kernel<<<BB * TT, 256, 0, stream>>>(trg, ln1g, ln1b, xn);
    gemm_kernel<<<gD, blkG, 0, stream>>>(xn, sa_qW, sa_qb, qb, nullptr, BB*TT, DD, DD, TT, ISQ, 0, 1);
    gemm_kernel<<<gD, blkG, 0, stream>>>(xn, sa_kW, sa_kb, kb, nullptr, BB*TT, DD, DD, TT, 1.f, 0, 1);
    gemm_kernel<<<gD, blkG, 0, stream>>>(xn, sa_vW, sa_vb, vb, nullptr, BB*TT, DD, DD, TT, 1.f, 0, 2);
    {
        int total = BB * HH * TT * RELSZ;
        relscore_kernel<<<(total + 255) / 256, 256, 0, stream>>>(qb, rel_k, Rr, total);
    }
    flash_kernel<true><<<dim3(TT / 64, BB * HH), blkF, 0, stream>>>(
        qb, kb, vb, trg_mask, 1, Rr, rel_v, ctx, nullptr, nullptr, TT, TT);
    gemm_kernel<<<gD, blkG, 0, stream>>>(ctx, sa_oW, sa_ob, x, trg, BB*TT, DD, DD, TT, 1.f, 0, 0);

    // ===== gated (AST) cross attention =====
    ln_kernel<<<BB * TT, 256, 0, stream>>>(x, ln2g, ln2b, xn);
    gemm_kernel<<<gD, blkG, 0, stream>>>(xn, ga_qW, ga_qb, qb, nullptr, BB*TT, DD, DD, TT, ISQ, 0, 1);
    {
        const dim3 gN(DD / 128, (BB * NN) / 128);
        gemm_kernel<<<gN, blkG, 0, stream>>>(ast, ga_kW, ga_kb, kb, nullptr, BB*NN, DD, DD, NN, 1.f, 0, 1);
        gemm_kernel<<<gN, blkG, 0, stream>>>(ast, ga_vW, ga_vb, vb, nullptr, BB*NN, DD, DD, NN, 1.f, 0, 2);
    }
    flash_kernel<false><<<dim3(TT / 64, BB * HH), blkF, 0, stream>>>(
        qb, kb, vb, node_mask, 0, nullptr, nullptr, ctx, nullptr, nullptr, TT, NN);
    gemm_kernel<<<gD, blkG, 0, stream>>>(ctx, ga_oW, ga_ob, x, x, BB*TT, DD, DD, TT, 1.f, 0, 0);

    // ===== code cross attention =====
    ln_kernel<<<BB * TT, 256, 0, stream>>>(x, ln3g, ln3b, xn);
    gemm_kernel<<<gD, blkG, 0, stream>>>(xn, ca_qW, ca_qb, qb, nullptr, BB*TT, DD, DD, TT, ISQ, 0, 1);
    {
        const dim3 gS(DD / 128, (BB * SS) / 128);
        gemm_kernel<<<gS, blkG, 0, stream>>>(code, ca_kW, ca_kb, kb, nullptr, BB*SS, DD, DD, SS, 1.f, 0, 1);
        gemm_kernel<<<gS, blkG, 0, stream>>>(code, ca_vW, ca_vb, vb, nullptr, BB*SS, DD, DD, SS, 1.f, 0, 2);
    }
    flash_kernel<false><<<dim3(TT / 64, BB * HH), blkF, 0, stream>>>(
        qb, kb, vb, src_mask, 0, nullptr, nullptr, ctx, mca, lca, TT, SS);
    ca_probs_kernel<<<dim3(SS / 128, TT / 16, BB), blkF, 0, stream>>>(
        qb, kb, src_mask, mca, lca, caw, TT, SS);
    gemm_kernel<<<gD, blkG, 0, stream>>>(ctx, ca_oW, ca_ob, x, x, BB*TT, DD, DD, TT, 1.f, 0, 0);

    // ===== FFN =====
    ln_kernel<<<BB * TT, 256, 0, stream>>>(x, ffg, ffb, xn);
    gemm_kernel<<<dim3(FF_ / 128, (BB * TT) / 128), blkG, 0, stream>>>(
        xn, ffW1, ffb1, ffh, nullptr, BB*TT, FF_, DD, TT, 1.f, 1, 0);
    gemm_kernel<<<gD, blkG, 0, stream>>>(
        ffh, ffW2, ffb2, out, x, BB*TT, DD, FF_, TT, 1.f, 0, 0);
}
